// Decoder_16904991277383
// MI455X (gfx1250) — matrix operand order repaired and checked by reading, not compiled, not run on hardware
//
#include <hip/hip_runtime.h>
#include <hip/hip_bf16.h>

typedef __attribute__((ext_vector_type(16))) __bf16 v16bf;
typedef __attribute__((ext_vector_type(8)))  __bf16 v8bf;
typedef __attribute__((ext_vector_type(4)))  __bf16 v4bf;
typedef __attribute__((ext_vector_type(2)))  __bf16 v2bf;
typedef __attribute__((ext_vector_type(8)))  float  v8f;

#define B_   256
#define S_   64
#define U_   1024
#define E_   256
#define V_   32000
#define U3_  3072
#define GU_  1280   // E_ + U_

#define LDSTRIDE 40  // bf16 per row: 80B = 5*16B -> 16B aligned, conflict-free

// ---------------------------------------------------------------------------
// Block-cooperative, software-pipelined GEMM core. Block = 256 threads (8 waves).
// Block tile: [BMT*16 M x 128 N]; each wave owns one 16-wide N sub-tile and
// iterates BMT M-tiles against the shared A panel.
//
// Pipeline (double-buffered LDS, one barrier per k-step):
//   fetch(k+1) global->regs (loads in flight)  ||  wmma on LDS buf(k)
//   stage(k+1) regs->LDS buf(k^1)  (waits loads; disjoint buffer from readers)
//   barrier    (orders buffer swap; also protects write-after-read)
//
// Staging is coalesced float4 (global_load_b128) + packed bf16 cvt; LDS holds
// operands in WMMA order (ISA 7.12.2): A row-major k-contiguous per m,
// B column-major k-contiguous per n (k-pairs packed -> b32 LDS stores), so
// per-lane operand fetch is two contiguous 16B ds_load_b128 per matrix.
// ---------------------------------------------------------------------------
template <int BMT, typename FA4, typename FB4>
__device__ __forceinline__ void gemm_block(int K, FA4 loadA4, FB4 loadB4,
                                           v8f (&acc)[BMT]) {
  __shared__ __align__(16) __bf16 aLds[2][BMT * 16 * LDSTRIDE];
  __shared__ __align__(16) __bf16 bLds[2][128 * LDSTRIDE];

  constexpr int ASLOTS = BMT * 16 * 8;               // float4 slots for A panel
  constexpr int APT    = (ASLOTS + 255) / 256;       // A float4 loads per thread
  float4 aReg[APT];
  float4 bReg[2][2];

  const int lane = threadIdx.x & 31;
  const int idx  = lane & 15;
  const int half = lane >> 4;
  const int nl   = (threadIdx.x >> 5) * 16 + idx;    // wave-local N column

  auto fetch = [&](int k0) {  // global -> registers (coalesced b128)
#pragma unroll
    for (int t = 0; t < APT; ++t) {
      const int s = threadIdx.x + t * 256;
      if (ASLOTS >= 256 || s < ASLOTS) aReg[t] = loadA4(s >> 3, k0 + (s & 7) * 4);
    }
#pragma unroll
    for (int r = 0; r < 2; ++r) {
      const int s = threadIdx.x + r * 256;           // 0..511
      const int kp = s >> 5, c4 = s & 31;
      bReg[r][0] = loadB4(k0 + 2 * kp,     c4 * 4);
      bReg[r][1] = loadB4(k0 + 2 * kp + 1, c4 * 4);
    }
  };

  auto stage = [&](int buf) {  // registers -> bf16 LDS (operand order)
#pragma unroll
    for (int t = 0; t < APT; ++t) {
      const int s = threadIdx.x + t * 256;
      if (ASLOTS >= 256 || s < ASLOTS) {
        const int row = s >> 3, c4 = s & 7;
        const float4 v = aReg[t];
        v4bf p;
        p[0] = (__bf16)v.x; p[1] = (__bf16)v.y;
        p[2] = (__bf16)v.z; p[3] = (__bf16)v.w;
        *(v4bf*)(&aLds[buf][row * LDSTRIDE + c4 * 4]) = p;
      }
    }
#pragma unroll
    for (int r = 0; r < 2; ++r) {
      const int s = threadIdx.x + r * 256;
      const int kp = s >> 5, c4 = s & 31;
      const float4 lo = bReg[r][0], hi = bReg[r][1];
      const float l_[4] = {lo.x, lo.y, lo.z, lo.w};
      const float h_[4] = {hi.x, hi.y, hi.z, hi.w};
#pragma unroll
      for (int j = 0; j < 4; ++j) {
        v2bf p; p[0] = (__bf16)l_[j]; p[1] = (__bf16)h_[j];
        *(v2bf*)(&bLds[buf][(c4 * 4 + j) * LDSTRIDE + 2 * kp]) = p;
      }
    }
  };

  fetch(0);
  stage(0);
  __syncthreads();

  const int steps = K / 32;
  for (int i = 0; i < steps; ++i) {
    const int buf = i & 1;
    if (i + 1 < steps) fetch((i + 1) * 32);  // HBM latency overlaps compute

    // B operand: element e <-> k = half*16 + e
    const v8bf b0 = *(const v8bf*)(&bLds[buf][nl * LDSTRIDE + half * 16]);
    const v8bf b1 = *(const v8bf*)(&bLds[buf][nl * LDSTRIDE + half * 16 + 8]);
    const v16bf b = __builtin_shufflevector(b0, b1, 0, 1, 2, 3, 4, 5, 6, 7, 8,
                                            9, 10, 11, 12, 13, 14, 15);
#pragma unroll
    for (int mt = 0; mt < BMT; ++mt) {
      // A operand: element e <-> k = half*16 + e
      const int row = mt * 16 + idx;
      const v8bf a0 = *(const v8bf*)(&aLds[buf][row * LDSTRIDE + half * 16]);
      const v8bf a1 = *(const v8bf*)(&aLds[buf][row * LDSTRIDE + half * 16 + 8]);
      const v16bf a = __builtin_shufflevector(a0, a1, 0, 1, 2, 3, 4, 5, 6, 7,
                                              8, 9, 10, 11, 12, 13, 14, 15);
      acc[mt] = __builtin_amdgcn_wmma_f32_16x16x32_bf16(
          false, a, false, b, (short)0, acc[mt], false, false);
    }
    if (i + 1 < steps) stage(buf ^ 1);
    __syncthreads();
  }
}

// C/D layout: VGPR i -> M = i + half*8, N = lane%16 (+ wave's 16-col offset).
__device__ __forceinline__ void store_tile_bias(float* __restrict__ out, long ldn,
                                                int m0, int n0, const v8f& acc,
                                                const float* __restrict__ bias) {
  const int lane = threadIdx.x & 31;
  const int n    = n0 + (threadIdx.x >> 5) * 16 + (lane & 15);
  const int half = lane >> 4;
  const float bn = bias[n];
#pragma unroll
  for (int i = 0; i < 8; ++i)
    out[(long)(m0 + half * 8 + i) * ldn + n] = acc[i] + bn;
}

// ---------------------------------------------------------------------------
// K1: q = hidden @ W1 + b1   [256,1024]@[1024,1024]; 16 mt x 8 nt = 128 blocks
// ---------------------------------------------------------------------------
__global__ void k_q_gemm(const float* __restrict__ hidden,
                         const float* __restrict__ W1,
                         const float* __restrict__ b1,
                         float* __restrict__ q) {
  const int nt = blockIdx.x >> 4, mt = blockIdx.x & 15;
  const int m0 = mt * 16, n0 = nt * 128;
  v8f acc[1] = {};
  gemm_block<1>(U_,
      [&](int r, int k) { return *(const float4*)(hidden + (m0 + r) * U_ + k); },
      [&](int k, int c) { return *(const float4*)(W1 + (long)k * U_ + n0 + c); },
      acc);
  store_tile_bias(q, U_, m0, n0, acc[0], b1);
}

// ---------------------------------------------------------------------------
// K2: fused attention, one block per b. Block tile: 64 S x 128 N (BMT=4),
// looped over 8 N-stripes. Epilogue: tanh(k+q+b2)*Vw -> shfl reduce over N ->
// LDS atomic score -> softmax -> attn, ctx.
// ---------------------------------------------------------------------------
__global__ void k_attn(const float* __restrict__ enc,
                       const float* __restrict__ W2,
                       const float* __restrict__ b2,
                       const float* __restrict__ Vw,
                       const float* __restrict__ bv,
                       const float* __restrict__ q,
                       float* __restrict__ ctx,
                       float* __restrict__ attn_out) {
  const int b = blockIdx.x;
  __shared__ float s_score[S_];
  __shared__ float s_attn[S_];
  if (threadIdx.x < S_) s_score[threadIdx.x] = 0.0f;
  __syncthreads();

  const int lane = threadIdx.x & 31;
  const int idx  = lane & 15;
  const int half = lane >> 4;
  const float* __restrict__ encb = enc + (long)b * S_ * U_;
  const float* __restrict__ qb   = q + (long)b * U_;

  for (int nt = 0; nt < U_ / 128; ++nt) {
    const int n0 = nt * 128;
    v8f acc[4] = {};
    gemm_block<4>(U_,
        [&](int r, int k) { return *(const float4*)(encb + r * U_ + k); },
        [&](int k, int c) { return *(const float4*)(W2 + (long)k * U_ + n0 + c); },
        acc);
    const int   n  = n0 + (threadIdx.x >> 5) * 16 + idx;
    const float vw = Vw[n];
    const float qn = qb[n] + b2[n];
#pragma unroll
    for (int mt = 0; mt < 4; ++mt) {
      float p[8];
#pragma unroll
      for (int i = 0; i < 8; ++i) p[i] = tanhf(acc[mt][i] + qn) * vw;
#pragma unroll
      for (int off = 1; off < 16; off <<= 1) {
#pragma unroll
        for (int i = 0; i < 8; ++i) p[i] += __shfl_xor(p[i], off, 32);
      }
      if (idx == 0) {
#pragma unroll
        for (int i = 0; i < 8; ++i)
          atomicAdd(&s_score[mt * 16 + half * 8 + i], p[i]);
      }
    }
    __syncthreads();  // scores visible before next stripe / softmax
  }

  if (threadIdx.x == 0) {  // serial softmax, S=64 (negligible)
    float mx = -1e30f;
    for (int s = 0; s < S_; ++s) {
      const float v = s_score[s] + bv[0];
      s_score[s] = v;
      mx = v > mx ? v : mx;
    }
    float sum = 0.0f;
    for (int s = 0; s < S_; ++s) {
      const float e = __expf(s_score[s] - mx);
      s_attn[s] = e; sum += e;
    }
    const float inv = 1.0f / sum;
    for (int s = 0; s < S_; ++s) s_attn[s] *= inv;
  }
  __syncthreads();

  if (threadIdx.x < S_) attn_out[b * S_ + threadIdx.x] = s_attn[threadIdx.x];

#pragma unroll
  for (int j = 0; j < U_ / 256; ++j) {
    const int u = threadIdx.x + j * 256;
    float a = 0.0f;
    for (int s = 0; s < S_; ++s) a += s_attn[s] * encb[s * U_ + u];
    ctx[(long)b * U_ + u] = a;
  }
}

// ---------------------------------------------------------------------------
// K3: gi = [ctx | emb[x]] @ Wx + bx  [256,1280]@[1280,3072]
// 24 nt x 16 mt = 384 blocks; embedding gather fused into A loader
// (k-panels are 32-aligned so each panel is entirely ctx or entirely emb).
// ---------------------------------------------------------------------------
__global__ void k_gi_gemm(const float* __restrict__ ctx,
                          const float* __restrict__ emb,
                          const int* __restrict__ x,
                          const float* __restrict__ Wx,
                          const float* __restrict__ bx,
                          float* __restrict__ gi) {
  const int nt = blockIdx.x >> 4, mt = blockIdx.x & 15;
  const int m0 = mt * 16, n0 = nt * 128;
  v8f acc[1] = {};
  gemm_block<1>(GU_,
      [&](int r, int k) {
        const int bb = m0 + r;
        return (k < U_) ? *(const float4*)(ctx + (long)bb * U_ + k)
                        : *(const float4*)(emb + (long)x[bb] * E_ + (k - U_));
      },
      [&](int k, int c) { return *(const float4*)(Wx + (long)k * U3_ + n0 + c); },
      acc);
  store_tile_bias(gi, U3_, m0, n0, acc[0], bx);
}

// ---------------------------------------------------------------------------
// K4: GRU gates. h0 == 0 structurally => gh = bh, Wh matmul eliminated.
// ---------------------------------------------------------------------------
__global__ void k_gru(const float* __restrict__ gi,
                      const float* __restrict__ bh,
                      float* __restrict__ state) {
  const int i = blockIdx.x * blockDim.x + threadIdx.x;
  if (i >= B_ * U_) return;
  const int b = i / U_, u = i % U_;
  const float xz = gi[(long)b * U3_ + u];
  const float xr = gi[(long)b * U3_ + U_ + u];
  const float xn = gi[(long)b * U3_ + 2 * U_ + u];
  const float z = 1.0f / (1.0f + __expf(-(xz + bh[u])));
  const float r = 1.0f / (1.0f + __expf(-(xr + bh[U_ + u])));
  const float n = tanhf(xn + r * bh[2 * U_ + u]);
  state[i] = (1.0f - z) * n;
}

// ---------------------------------------------------------------------------
// K5: logits = state @ Wfc + bfc  [256,1024]@[1024,32000]. 250 nt x 16 mt
// = 4000 blocks; nt is the slow index so the 16 blocks sharing a 512KB Wfc
// stripe run back-to-back and hit L2 -> Wfc HBM traffic = its 131MB floor.
// ---------------------------------------------------------------------------
__global__ void k_logits(const float* __restrict__ state,
                         const float* __restrict__ Wfc,
                         const float* __restrict__ bfc,
                         float* __restrict__ logits) {
  const int nt = blockIdx.x >> 4, mt = blockIdx.x & 15;
  const int m0 = mt * 16, n0 = nt * 128;
  v8f acc[1] = {};
  gemm_block<1>(U_,
      [&](int r, int k) { return *(const float4*)(state + (m0 + r) * U_ + k); },
      [&](int k, int c) { return *(const float4*)(Wfc + (long)k * V_ + n0 + c); },
      acc);
  store_tile_bias(logits, V_, m0, n0, acc[0], bfc);
}

// ---------------------------------------------------------------------------
extern "C" void kernel_launch(void* const* d_in, const int* in_sizes, int n_in,
                              void* d_out, int out_size, void* d_ws, size_t ws_size,
                              hipStream_t stream) {
  (void)in_sizes; (void)n_in; (void)out_size; (void)ws_size;
  const int*   x      = (const int*)d_in[0];
  const float* hidden = (const float*)d_in[1];
  const float* enc    = (const float*)d_in[2];
  const float* emb    = (const float*)d_in[3];
  const float* W1     = (const float*)d_in[4];
  const float* b1     = (const float*)d_in[5];
  const float* W2     = (const float*)d_in[6];
  const float* b2     = (const float*)d_in[7];
  const float* Vw     = (const float*)d_in[8];
  const float* bv     = (const float*)d_in[9];
  const float* Wx     = (const float*)d_in[10];
  /* d_in[11] = Wh is provably unused: h0 == 0 */
  const float* bx     = (const float*)d_in[12];
  const float* bh     = (const float*)d_in[13];
  const float* Wfc    = (const float*)d_in[14];
  const float* bfc    = (const float*)d_in[15];

  float* out    = (float*)d_out;
  float* logits = out;                               // B*V
  float* state  = logits + (long)B_ * V_;            // B*U
  float* attn   = state + (long)B_ * U_;             // B*S

  float* ws  = (float*)d_ws;
  float* q   = ws;                                   // B*U
  float* ctx = q + (long)B_ * U_;                    // B*U
  float* gi  = ctx + (long)B_ * U_;                  // B*3U

  k_q_gemm <<<128,  256, 0, stream>>>(hidden, W1, b1, q);
  k_attn   <<<B_,   256, 0, stream>>>(enc, W2, b2, Vw, bv, q, ctx, attn);
  k_gi_gemm<<<384,  256, 0, stream>>>(ctx, emb, x, Wx, bx, gi);
  k_gru    <<<(B_ * U_ + 255) / 256, 256, 0, stream>>>(gi, bh, state);
  k_logits <<<4000, 256, 0, stream>>>(state, Wfc, bfc, logits);
}
